// TransformerBlock_43963285242664
// MI455X (gfx1250) — compile-verified, matrix operand order checked by reading, not run on hardware
//
#include <hip/hip_runtime.h>
#include <hip/hip_bf16.h>

// ---------------------------------------------------------------- types ----
typedef __attribute__((ext_vector_type(16))) __bf16        v16bf;
typedef __attribute__((ext_vector_type(8)))  float         v8f;
typedef __attribute__((ext_vector_type(4)))  float         v4f;
typedef __attribute__((ext_vector_type(4)))  unsigned short v4u16;
typedef unsigned short us;

#define LQ   2048
#define DIM  1024
#define NH   16
#define HD   64
#define MLPD 8192
#define MLPH 4096

__device__ __forceinline__ us f2bf(float f) {
  unsigned int u = __builtin_bit_cast(unsigned int, f);
  u += 0x7FFFu + ((u >> 16) & 1u);        // round-to-nearest-even
  return (us)(u >> 16);
}

__device__ __forceinline__ v8f wmma_bf16(v16bf a, v16bf b, v8f c) {
  return __builtin_amdgcn_wmma_f32_16x16x32_bf16(false, a, false, b,
                                                 (short)0, c, false, false);
}

// Generic pointer low 32 bits == LDS byte address (ISA 10.2 aperture rule).
__device__ __forceinline__ unsigned lds_addr32(const void* p) {
  return (unsigned)(size_t)p;
}

// 32B async copy global->LDS (two b128 lanes-worth per thread), ASYNCcnt path.
__device__ __forceinline__ void async_cp32(const us* base, unsigned gbyte,
                                           unsigned lbyte) {
  asm volatile("global_load_async_to_lds_b128 %0, %1, %2"
               :: "v"(lbyte), "v"(gbyte), "s"(base) : "memory");
  asm volatile("global_load_async_to_lds_b128 %0, %1, %2"
               :: "v"(lbyte + 16u), "v"(gbyte + 16u), "s"(base) : "memory");
}
__device__ __forceinline__ void wait_async0() {
  asm volatile("s_wait_asynccnt 0" ::: "memory");
}

// ------------------------------------------------- weight f32 -> bf16 -----
__global__ __launch_bounds__(256) void cvt_bf16_kernel(
    const float* __restrict__ in, us* __restrict__ out, int n4) {
  int i = blockIdx.x * 256 + threadIdx.x;
  if (i >= n4) return;
  v4f v = *(const v4f*)(in + (size_t)i * 4);
  v4u16 o;
  o[0] = f2bf(v[0]); o[1] = f2bf(v[1]); o[2] = f2bf(v[2]); o[3] = f2bf(v[3]);
  *(v4u16*)(out + (size_t)i * 4) = o;
}

// -------------------------------------------------------- LayerNorm -------
__global__ __launch_bounds__(256) void ln_kernel(
    const float* __restrict__ x, us* __restrict__ out) {
  __shared__ float s_sum[8], s_sq[8];
  int row = blockIdx.x, tid = threadIdx.x;
  v4f v = *(const v4f*)(x + (size_t)row * DIM + tid * 4);
  float s = v[0] + v[1] + v[2] + v[3];
  float q = v[0]*v[0] + v[1]*v[1] + v[2]*v[2] + v[3]*v[3];
  #pragma unroll
  for (int m = 16; m >= 1; m >>= 1) {
    s += __shfl_xor(s, m);
    q += __shfl_xor(q, m);
  }
  if ((tid & 31) == 0) { s_sum[tid >> 5] = s; s_sq[tid >> 5] = q; }
  __syncthreads();
  float ts = 0.f, tq = 0.f;
  #pragma unroll
  for (int w = 0; w < 8; ++w) { ts += s_sum[w]; tq += s_sq[w]; }
  float mean = ts * (1.0f / DIM);
  float var  = tq * (1.0f / DIM) - mean * mean;
  float rstd = rsqrtf(var + 1e-5f);
  v4u16 o;
  #pragma unroll
  for (int i = 0; i < 4; ++i) o[i] = f2bf((v[i] - mean) * rstd);
  *(v4u16*)(out + (size_t)row * DIM + tid * 4) = o;
}

// ------------------------------------------- generic NT GEMM (bf16 WMMA) --
// 128x128x32 tiles, 8 waves (2x4), wave tile 64x32, async double-buffered LDS.
template<int EPI>
__global__ __launch_bounds__(256) void gemm_nt_kernel(
    const us* __restrict__ A, const us* __restrict__ B,
    const float* __restrict__ RES, float* __restrict__ C,
    int M, int N, int K) {
  __shared__ us lsA[2][128 * 48];
  __shared__ us lsB[2][128 * 48];
  const unsigned BUF = 128 * 48 * 2;
  int tid = threadIdx.x;
  int lane = tid & 31, wave = tid >> 5;
  int wm = wave >> 2, wn = wave & 3;
  int lhalf = lane >> 4, l16 = lane & 15;
  int m0 = blockIdx.y * 128, n0 = blockIdx.x * 128;

  v8f acc[4][2];
  #pragma unroll
  for (int i = 0; i < 4; ++i)
    #pragma unroll
    for (int j = 0; j < 2; ++j) acc[i][j] = {};

  int row = tid >> 1, seg = tid & 1;
  unsigned gA = (unsigned)((((size_t)(m0 + row)) * K + seg * 16) * 2);
  unsigned gB = (unsigned)((((size_t)(n0 + row)) * K + seg * 16) * 2);
  unsigned lOff = (unsigned)((row * 48 + seg * 16) * 2);
  unsigned laBase = lds_addr32(&lsA[0][0]);
  unsigned lbBase = lds_addr32(&lsB[0][0]);

  async_cp32(A, gA, laBase + lOff);
  async_cp32(B, gB, lbBase + lOff);
  wait_async0();
  __syncthreads();

  int nk = K / 32;
  for (int it = 0; it < nk; ++it) {
    int cur = it & 1;
    if (it + 1 < nk) {
      int nxt = cur ^ 1;
      unsigned kb = (unsigned)((it + 1) * 64);   // 32 elems * 2B
      async_cp32(A, gA + kb, laBase + nxt * BUF + lOff);
      async_cp32(B, gB + kb, lbBase + nxt * BUF + lOff);
    }
    const us* cA = &lsA[cur][0];
    const us* cB = &lsB[cur][0];
    v16bf aF[4], bF[2];
    #pragma unroll
    for (int i = 0; i < 4; ++i)
      aF[i] = *(const v16bf*)(cA + (wm * 64 + i * 16 + l16) * 48 + lhalf * 16);
    #pragma unroll
    for (int j = 0; j < 2; ++j)
      bF[j] = *(const v16bf*)(cB + (wn * 32 + j * 16 + l16) * 48 + lhalf * 16);
    #pragma unroll
    for (int i = 0; i < 4; ++i)
      #pragma unroll
      for (int j = 0; j < 2; ++j)
        acc[i][j] = wmma_bf16(aF[i], bF[j], acc[i][j]);
    wait_async0();
    __syncthreads();
  }
  #pragma unroll
  for (int i = 0; i < 4; ++i)
    #pragma unroll
    for (int j = 0; j < 2; ++j)
      #pragma unroll
      for (int r = 0; r < 8; ++r) {
        int m = m0 + wm * 64 + i * 16 + r + lhalf * 8;
        int n = n0 + wn * 32 + j * 16 + l16;
        size_t o = (size_t)m * N + n;
        float v = acc[i][j][r];
        if (EPI == 1) v += RES[o];
        C[o] = v;
      }
}

// ------------------------------- FFN1: dual-tile GEMM + fused SiLU gate ---
__global__ __launch_bounds__(256) void ffn1_gate_kernel(
    const us* __restrict__ A, const us* __restrict__ B,
    us* __restrict__ ACT, int M, int Nh, int K) {
  __shared__ us lsA[2][128 * 48];
  __shared__ us lsB1[2][128 * 48];
  __shared__ us lsB2[2][128 * 48];
  const unsigned BUF = 128 * 48 * 2;
  int tid = threadIdx.x;
  int lane = tid & 31, wave = tid >> 5;
  int wm = wave >> 2, wn = wave & 3;
  int lhalf = lane >> 4, l16 = lane & 15;
  int m0 = blockIdx.y * 128, n0 = blockIdx.x * 128;

  v8f acc1[4][2], acc2[4][2];
  #pragma unroll
  for (int i = 0; i < 4; ++i)
    #pragma unroll
    for (int j = 0; j < 2; ++j) { acc1[i][j] = {}; acc2[i][j] = {}; }

  int row = tid >> 1, seg = tid & 1;
  unsigned gA  = (unsigned)((((size_t)(m0 + row)) * K + seg * 16) * 2);
  unsigned gB1 = (unsigned)((((size_t)(n0 + row)) * K + seg * 16) * 2);
  unsigned gB2 = (unsigned)((((size_t)(n0 + row + Nh)) * K + seg * 16) * 2);
  unsigned lOff = (unsigned)((row * 48 + seg * 16) * 2);
  unsigned laB  = lds_addr32(&lsA[0][0]);
  unsigned lb1B = lds_addr32(&lsB1[0][0]);
  unsigned lb2B = lds_addr32(&lsB2[0][0]);

  async_cp32(A, gA, laB + lOff);
  async_cp32(B, gB1, lb1B + lOff);
  async_cp32(B, gB2, lb2B + lOff);
  wait_async0();
  __syncthreads();

  int nk = K / 32;
  for (int it = 0; it < nk; ++it) {
    int cur = it & 1;
    if (it + 1 < nk) {
      int nxt = cur ^ 1;
      unsigned kb = (unsigned)((it + 1) * 64);
      async_cp32(A, gA + kb, laB + nxt * BUF + lOff);
      async_cp32(B, gB1 + kb, lb1B + nxt * BUF + lOff);
      async_cp32(B, gB2 + kb, lb2B + nxt * BUF + lOff);
    }
    const us* cA = &lsA[cur][0];
    const us* c1 = &lsB1[cur][0];
    const us* c2 = &lsB2[cur][0];
    v16bf aF[4], b1F[2], b2F[2];
    #pragma unroll
    for (int i = 0; i < 4; ++i)
      aF[i] = *(const v16bf*)(cA + (wm * 64 + i * 16 + l16) * 48 + lhalf * 16);
    #pragma unroll
    for (int j = 0; j < 2; ++j) {
      b1F[j] = *(const v16bf*)(c1 + (wn * 32 + j * 16 + l16) * 48 + lhalf * 16);
      b2F[j] = *(const v16bf*)(c2 + (wn * 32 + j * 16 + l16) * 48 + lhalf * 16);
    }
    #pragma unroll
    for (int i = 0; i < 4; ++i)
      #pragma unroll
      for (int j = 0; j < 2; ++j) {
        acc1[i][j] = wmma_bf16(aF[i], b1F[j], acc1[i][j]);
        acc2[i][j] = wmma_bf16(aF[i], b2F[j], acc2[i][j]);
      }
    wait_async0();
    __syncthreads();
  }
  #pragma unroll
  for (int i = 0; i < 4; ++i)
    #pragma unroll
    for (int j = 0; j < 2; ++j)
      #pragma unroll
      for (int r = 0; r < 8; ++r) {
        int m = m0 + wm * 64 + i * 16 + r + lhalf * 8;
        int n = n0 + wn * 32 + j * 16 + l16;
        float x1 = acc1[i][j][r];
        float x2 = acc2[i][j][r];
        // silu(x2)*x1 with fast v_rcp_f32 instead of IEEE divide
        float g = x2 * x1 * __builtin_amdgcn_rcpf(1.0f + __expf(-x2));
        ACT[(size_t)m * Nh + n] = f2bf(g);
      }
}

// ---------------------------------------- RoPE + head split (QKV -> Q/K/V)
__global__ __launch_bounds__(256) void rope_split_kernel(
    const float* __restrict__ qkv, us* __restrict__ Qh,
    us* __restrict__ Kh, us* __restrict__ Vt) {
  int idx = blockIdx.x * 256 + threadIdx.x;
  int l  = idx >> 10;
  int c  = idx & 1023;
  int h  = c >> 6;
  int hd = c & 63;
  int i  = hd & 31;
  float inv = __expf(-((float)i) * (9.210340371976184f / 32.0f));
  float ang = (float)l * inv;
  float cs = __cosf(ang), sn = __sinf(ang);
  size_t rowb = (size_t)l * (3 * DIM) + h * 64;
  float qa = qkv[rowb + i],        qb = qkv[rowb + 32 + i];
  float ka = qkv[rowb + DIM + i],  kb = qkv[rowb + DIM + 32 + i];
  float qv = (hd < 32) ? (qa * cs - qb * sn) : (qb * cs + qa * sn);
  float kv = (hd < 32) ? (ka * cs - kb * sn) : (kb * cs + ka * sn);
  float vv = qkv[rowb + 2 * DIM + hd];
  size_t o = ((size_t)h * LQ + l) * 64 + hd;
  Qh[o] = f2bf(qv * 0.125f);                   // fold HD^-0.5
  Kh[o] = f2bf(kv);
  Vt[((size_t)h * 64 + hd) * LQ + l] = f2bf(vv);
}

// --------------------------------------------- flash attention per head ---
__global__ __launch_bounds__(128) void attn_kernel(
    const us* __restrict__ Qh, const us* __restrict__ Kh,
    const us* __restrict__ Vt, us* __restrict__ O) {
  __shared__ us lsQ[64 * 80];
  __shared__ us lsK[64 * 80];
  __shared__ us lsVT[64 * 80];
  __shared__ us lsP[64 * 80];
  int tid = threadIdx.x;
  int lane = tid & 31, wave = tid >> 5;
  int lhalf = lane >> 4, l16 = lane & 15;
  int qb = blockIdx.x, head = blockIdx.y;

  int row = tid >> 1, seg = tid & 1;
  unsigned lTile = (unsigned)((row * 80 + seg * 32) * 2);
  const us* baseQ = Qh + (size_t)head * LQ * 64;
  const us* baseK = Kh + (size_t)head * LQ * 64;
  const us* baseV = Vt + (size_t)head * 64 * LQ;

  // async-load Q block (64 x 64)
  {
    unsigned g = (unsigned)(((qb * 64 + row) * 64 + seg * 32) * 2);
    unsigned l = lds_addr32(lsQ) + lTile;
    async_cp32(baseQ, g, l);
    async_cp32(baseQ, g + 32, l + 32);
  }
  wait_async0();
  __syncthreads();
  v16bf qF[2];
  #pragma unroll
  for (int s = 0; s < 2; ++s)
    qF[s] = *(const v16bf*)(lsQ + (wave * 16 + l16) * 80 + s * 32 + lhalf * 16);

  float mrow[8], lrow[8];
  v8f oacc[4];
  #pragma unroll
  for (int r = 0; r < 8; ++r) { mrow[r] = -1e30f; lrow[r] = 0.f; }
  #pragma unroll
  for (int t = 0; t < 4; ++t) oacc[t] = {};

  for (int kb = 0; kb <= qb; ++kb) {
    __syncthreads();
    {
      unsigned gk = (unsigned)(((kb * 64 + row) * 64 + seg * 32) * 2);
      unsigned lk = lds_addr32(lsK) + lTile;
      async_cp32(baseK, gk, lk);
      async_cp32(baseK, gk + 32, lk + 32);
      unsigned gv = (unsigned)((row * LQ + kb * 64 + seg * 32) * 2);
      unsigned lv = lds_addr32(lsVT) + lTile;
      async_cp32(baseV, gv, lv);
      async_cp32(baseV, gv + 32, lv + 32);
    }
    wait_async0();
    __syncthreads();

    // S = Q * K^T
    v8f S[4];
    #pragma unroll
    for (int nt = 0; nt < 4; ++nt) {
      S[nt] = {};
      #pragma unroll
      for (int s = 0; s < 2; ++s) {
        v16bf kF = *(const v16bf*)(lsK + (nt * 16 + l16) * 80 + s * 32 + lhalf * 16);
        S[nt] = wmma_bf16(qF[s], kF, S[nt]);
      }
    }
    if (kb == qb) {  // causal mask on diagonal block
      #pragma unroll
      for (int nt = 0; nt < 4; ++nt)
        #pragma unroll
        for (int r = 0; r < 8; ++r) {
          int qrow = wave * 16 + r + lhalf * 8;
          int kcol = nt * 16 + l16;
          if (kcol > qrow) S[nt][r] = -1e30f;
        }
    }
    // online softmax; stage P in per-wave LDS patch
    #pragma unroll
    for (int r = 0; r < 8; ++r) {
      float smax = fmaxf(fmaxf(S[0][r], S[1][r]), fmaxf(S[2][r], S[3][r]));
      #pragma unroll
      for (int m = 8; m >= 1; m >>= 1) smax = fmaxf(smax, __shfl_xor(smax, m));
      float mn = fmaxf(mrow[r], smax);
      float alpha = __expf(mrow[r] - mn);
      float psum = 0.f;
      int mloc = r + lhalf * 8;
      #pragma unroll
      for (int nt = 0; nt < 4; ++nt) {
        float p = __expf(S[nt][r] - mn);
        psum += p;
        lsP[(wave * 16 + mloc) * 80 + nt * 16 + l16] = f2bf(p);
      }
      #pragma unroll
      for (int m = 8; m >= 1; m >>= 1) psum += __shfl_xor(psum, m);
      lrow[r] = lrow[r] * alpha + psum;
      mrow[r] = mn;
      #pragma unroll
      for (int t = 0; t < 4; ++t) oacc[t][r] *= alpha;
    }
    // O += P * V
    v16bf pF[2];
    #pragma unroll
    for (int s = 0; s < 2; ++s)
      pF[s] = *(const v16bf*)(lsP + (wave * 16 + l16) * 80 + s * 32 + lhalf * 16);
    #pragma unroll
    for (int t = 0; t < 4; ++t)
      #pragma unroll
      for (int s = 0; s < 2; ++s) {
        v16bf vF = *(const v16bf*)(lsVT + (t * 16 + l16) * 80 + s * 32 + lhalf * 16);
        oacc[t] = wmma_bf16(pF[s], vF, oacc[t]);
      }
  }
  #pragma unroll
  for (int t = 0; t < 4; ++t)
    #pragma unroll
    for (int r = 0; r < 8; ++r) {
      int orow = qb * 64 + wave * 16 + r + lhalf * 8;
      int col = head * 64 + t * 16 + l16;
      float invl = __builtin_amdgcn_rcpf(lrow[r]);
      O[(size_t)orow * DIM + col] = f2bf(oacc[t][r] * invl);
    }
}

// ------------------------------------------------------------- launcher ---
extern "C" void kernel_launch(void* const* d_in, const int* in_sizes, int n_in,
                              void* d_out, int out_size, void* d_ws, size_t ws_size,
                              hipStream_t stream) {
  const float* x      = (const float*)d_in[0];
  const float* W_attn = (const float*)d_in[2];
  const float* W_out  = (const float*)d_in[3];
  const float* W_ffp  = (const float*)d_in[4];
  const float* W_ffo  = (const float*)d_in[5];
  float* out = (float*)d_out;

  unsigned char* w = (unsigned char*)d_ws;
  size_t off = 0;
  auto take = [&](size_t bytes) {
    void* p = w + off;
    off = (off + bytes + 255) & ~(size_t)255;
    return p;
  };
  us*    wA   = (us*)take((size_t)3 * DIM * DIM * 2);
  us*    wO   = (us*)take((size_t)DIM * DIM * 2);
  us*    wP   = (us*)take((size_t)MLPD * DIM * 2);
  us*    wF   = (us*)take((size_t)DIM * MLPH * 2);
  us*    xn   = (us*)take((size_t)LQ * DIM * 2);
  float* qkv  = (float*)take((size_t)LQ * 3 * DIM * 4);
  us*    QhB  = (us*)take((size_t)NH * LQ * HD * 2);
  us*    KhB  = (us*)take((size_t)NH * LQ * HD * 2);
  us*    VtB  = (us*)take((size_t)NH * HD * LQ * 2);
  us*    attn = (us*)take((size_t)LQ * DIM * 2);
  float* hbuf = (float*)take((size_t)LQ * DIM * 4);
  us*    hn   = (us*)take((size_t)LQ * DIM * 2);
  us*    act  = (us*)take((size_t)LQ * MLPH * 2);

  cvt_bf16_kernel<<<(3 * DIM * DIM / 4) / 256, 256, 0, stream>>>(W_attn, wA, 3 * DIM * DIM / 4);
  cvt_bf16_kernel<<<(DIM * DIM / 4) / 256, 256, 0, stream>>>(W_out, wO, DIM * DIM / 4);
  cvt_bf16_kernel<<<(MLPD * DIM / 4) / 256, 256, 0, stream>>>(W_ffp, wP, MLPD * DIM / 4);
  cvt_bf16_kernel<<<(DIM * MLPH / 4) / 256, 256, 0, stream>>>(W_ffo, wF, DIM * MLPH / 4);

  ln_kernel<<<LQ, 256, 0, stream>>>(x, xn);

  gemm_nt_kernel<0><<<dim3(3 * DIM / 128, LQ / 128), 256, 0, stream>>>(
      xn, wA, nullptr, qkv, LQ, 3 * DIM, DIM);

  rope_split_kernel<<<(LQ * DIM) / 256, 256, 0, stream>>>(qkv, QhB, KhB, VtB);

  attn_kernel<<<dim3(LQ / 64, NH), 128, 0, stream>>>(QhB, KhB, VtB, attn);

  gemm_nt_kernel<1><<<dim3(DIM / 128, LQ / 128), 256, 0, stream>>>(
      attn, wO, x, hbuf, LQ, DIM, DIM);

  ln_kernel<<<LQ, 256, 0, stream>>>(hbuf, hn);

  ffn1_gate_kernel<<<dim3(MLPH / 128, LQ / 128), 256, 0, stream>>>(
      hn, wP, act, LQ, MLPH, DIM);

  gemm_nt_kernel<1><<<dim3(DIM / 128, LQ / 128), 256, 0, stream>>>(
      act, wF, hbuf, out, LQ, DIM, MLPH);
}
